// LSTM_27934467293427
// MI455X (gfx1250) — matrix operand order repaired and checked by reading, not compiled, not run on hardware
//
#include <hip/hip_runtime.h>

typedef __attribute__((ext_vector_type(16))) _Float16 v16h;
typedef __attribute__((ext_vector_type(8)))  _Float16 v8h;
typedef __attribute__((ext_vector_type(8)))  float    v8f;
typedef __attribute__((ext_vector_type(4)))  unsigned v4u;
typedef __attribute__((ext_vector_type(8)))  int      v8i;
typedef __attribute__((ext_vector_type(4)))  int      v4i;

#define B_   256
#define S_   2048
#define E_   128
#define H_   64
#define G_   256      // 4*H
#define K_   13
#define H2_  2
#define NH_  (K_ * H2_)   // 26 head output columns
#define ROWS 16       // batch rows per workgroup
#define NWG  (B_ / ROWS)
#define XLD  132      // padded LDS row stride for x tile, f32 (128 + 4)
#define HLD16 72      // padded LDS row stride for h, f16 (64 + 8) -> 144B, 16B aligned
#define GLD  260      // padded LDS row stride for gates, f32 (256 + 4)
#define ZLD  28       // padded LDS row stride for head pre-activations
#define NHEAD (ROWS * NH_)   // 416 head items per WG

#if defined(__gfx1250__) && __has_builtin(__builtin_amdgcn_tensor_load_to_lds)
#define USE_TDM 1
#else
#define USE_TDM 0
#endif

__device__ __forceinline__ float sigmoidf_(float x) {
    return 1.0f / (1.0f + __expf(-x));
}

// D = A(16x32 f16) * B(32x16 f16) + C(16x16 f32)
__device__ __forceinline__ v8f wmma_f16(v16h a, v16h b, v8f c) {
    return __builtin_amdgcn_wmma_f32_16x16x32_f16(
        /*neg_a=*/false, a, /*neg_b=*/false, b,
        /*c_mod=*/(short)0, c, /*reuse_a=*/false, /*reuse_b=*/false);
}

// 16x32 A-fragment gather from f32 LDS (converted to f16).
// Contiguous K order (B-fragments match): lanes 0-15 carry M=0..15 with
// K in kb+0..15; lanes 16-31 same M, K = kb+16..31.
__device__ __forceinline__ v16h gather_a_f32(const float* __restrict__ src, int ld,
                                             int lane, int kbase) {
    const int m  = lane & 15;
    const int kh = kbase + ((lane >> 4) << 4);
    const float* p = src + m * ld + kh;
    float4 x0 = *(const float4*)(p + 0);
    float4 x1 = *(const float4*)(p + 4);
    float4 x2 = *(const float4*)(p + 8);
    float4 x3 = *(const float4*)(p + 12);
    v16h a;
    a[0]  = (_Float16)x0.x; a[1]  = (_Float16)x0.y;
    a[2]  = (_Float16)x0.z; a[3]  = (_Float16)x0.w;
    a[4]  = (_Float16)x1.x; a[5]  = (_Float16)x1.y;
    a[6]  = (_Float16)x1.z; a[7]  = (_Float16)x1.w;
    a[8]  = (_Float16)x2.x; a[9]  = (_Float16)x2.y;
    a[10] = (_Float16)x2.z; a[11] = (_Float16)x2.w;
    a[12] = (_Float16)x3.x; a[13] = (_Float16)x3.y;
    a[14] = (_Float16)x3.z; a[15] = (_Float16)x3.w;
    return a;
}

// 16x32 A-fragment gather from f16 LDS (no conversion, 2x ds_load_b128).
__device__ __forceinline__ v16h gather_a_f16(const _Float16* __restrict__ src, int ld,
                                             int lane, int kbase) {
    const int m  = lane & 15;
    const int kh = kbase + ((lane >> 4) << 4);
    const _Float16* p = src + m * ld + kh;
    v8h lo = *(const v8h*)(p);        // K = kh+0..7
    v8h hi = *(const v8h*)(p + 8);    // K = kh+8..15
    v16h a;
    #pragma unroll
    for (int j = 0; j < 8; ++j) { a[j] = lo[j]; a[8 + j] = hi[j]; }
    return a;
}

#if USE_TDM
// TDM 2D tile load: ROWS x 128 f32, row stride S_*E_ elements, into a padded
// LDS buffer (pad 4 dwords after every 128 dwords -> XLD=132 layout).
__device__ __forceinline__ void tdm_issue(unsigned lds_addr, const float* gptr) {
    unsigned long long ga = (unsigned long long)(size_t)gptr;
    v4u g0;
    g0[0] = 1u;                                   // count=1 valid descriptor
    g0[1] = lds_addr;                             // lds_addr
    g0[2] = (unsigned)ga;                         // global_addr[31:0]
    g0[3] = (unsigned)((ga >> 32) & 0x1FFFFFFull) // global_addr[56:32]
          | (2u << 30);                           // type=2 ("image")
    v8i g1;
    g1[0] = (2 << 16)                             // data_size = 4 bytes
          | (1 << 20)                             // pad_enable
          | (6 << 22)                             // pad_interval: every 128 dwords
          | (3 << 25);                            // pad_amount: 4 dwords
    g1[1] = (int)((unsigned)E_ << 16);            // tensor_dim0 = 128 (lo16)
    g1[2] = (int)((unsigned)ROWS << 16);          // tensor_dim1 = 16  (lo16)
    g1[3] = (int)((unsigned)E_ << 16);            // tile_dim0 = 128
    g1[4] = ROWS;                                 // tile_dim1 = 16, tile_dim2 = 0
    g1[5] = S_ * E_;                              // tensor_dim0_stride (lo32)
    g1[6] = 0;
    g1[7] = 0;
    v4i z4 = {0, 0, 0, 0};
#if __clang_major__ >= 23
    v8i z8 = {0, 0, 0, 0, 0, 0, 0, 0};
    __builtin_amdgcn_tensor_load_to_lds(g0, g1, z4, z4, z8, 0);
#else
    __builtin_amdgcn_tensor_load_to_lds(g0, g1, z4, z4, 0);
#endif
}
#endif

__global__ __launch_bounds__(256) void lstm_head_persist(
    const float* __restrict__ input,   // (B,S,E)
    const long long* __restrict__ lengths,
    const float* __restrict__ Wih,     // (4H,E)
    const float* __restrict__ Whh,     // (4H,H)
    const float* __restrict__ bih,     // (4H)
    const float* __restrict__ bhh,     // (4H)
    const float* __restrict__ hWih,    // (K,H2,H) == (26,64) row-major
    const float* __restrict__ hWhh,    // (K,H2,H2)
    const float* __restrict__ hbih,    // (K,H2)
    const float* __restrict__ hbhh,    // (K,H2)
    float* __restrict__ out)           // (K,B,H2)
{
    __shared__ __align__(16) float    xstage[2][ROWS * XLD];
    __shared__ __align__(16) float    g_s[ROWS * GLD];
    __shared__ __align__(16) _Float16 h16_s[ROWS * HLD16];
    __shared__ __align__(16) float    c_s[ROWS * H_];
    __shared__ float hk_s[ROWS * NH_];
    __shared__ float z_s[ROWS * ZLD];
    __shared__ float hR_s[NH_ * H2_];
    __shared__ float hb_s[NH_];
    __shared__ int   len_s[ROWS];

    const int tid   = threadIdx.x;
    const int lane  = tid & 31;
    const int wave  = tid >> 5;
    const int bbase = blockIdx.x * ROWS;

    // ---- one-time init: states, head params, lengths ----
    for (int i = tid; i < ROWS * HLD16; i += 256) h16_s[i] = (_Float16)0.0f;
    for (int i = tid; i < ROWS * H_; i += 256) c_s[i] = 0.0f;
    for (int i = tid; i < ROWS * NH_; i += 256) hk_s[i] = 0.0f;
    if (tid < NH_ * H2_) hR_s[tid] = hWhh[tid];
    if (tid < NH_) hb_s[tid] = hbih[tid] + hbhh[tid];
    if (tid < ROWS) len_s[tid] = (int)lengths[bbase + tid];

    // ---- one-time: f32 weights -> register-resident f16 B-fragments ----
    // Wave w owns gate columns [w*32, w*32+32): two 16-wide N tiles.
    // B-fragment (32x16, 16-bit): lane = n-in-tile (0..15) + 16*khalf,
    // element j -> K = kbase + 16*(lane>>4) + j.
    const int n0 = wave * 32;
    v16h wif[2][4];   // Wih^T fragments, 4 K-chunks over E=128
    v16h whf[2][2];   // Whh^T fragments, 2 K-chunks over H=64
    float gb[2];
    #pragma unroll
    for (int t2 = 0; t2 < 2; ++t2) {
        const int nn = n0 + t2 * 16 + (lane & 15);
        const int kb = (lane >> 4) << 4;
        #pragma unroll
        for (int kc = 0; kc < 4; ++kc) {
            const float* wp = Wih + (size_t)nn * E_ + kc * 32 + kb;
            #pragma unroll
            for (int j = 0; j < 16; ++j) wif[t2][kc][j] = (_Float16)wp[j];
        }
        #pragma unroll
        for (int kc = 0; kc < 2; ++kc) {
            const float* wp = Whh + (size_t)nn * H_ + kc * 32 + kb;
            #pragma unroll
            for (int j = 0; j < 16; ++j) whf[t2][kc][j] = (_Float16)wp[j];
        }
        gb[t2] = bih[nn] + bhh[nn];
    }

    // Head GEMM B-fragments on waves 0-1: z = h(16x64) . hW^T(64x26),
    // N tile = wave*16, columns >= 26 zero-filled.
    v16h hwf[2] = {};
    if (wave < 2) {
        const int ncol = wave * 16 + (lane & 15);
        const int kb = (lane >> 4) << 4;
        const bool valid = (ncol < NH_);
        const float* wp = hWih + (size_t)(valid ? ncol : 0) * H_ + kb;
        #pragma unroll
        for (int kc = 0; kc < 2; ++kc) {
            #pragma unroll
            for (int j = 0; j < 16; ++j)
                hwf[kc][j] = valid ? (_Float16)wp[kc * 32 + j] : (_Float16)0.0f;
        }
    }

    const float* xg = input + (size_t)bbase * S_ * E_;

#if USE_TDM
    if (wave == 0) {
        tdm_issue((unsigned)(size_t)&xstage[0][0], xg);   // prefetch t = 0
    }
#endif
    __syncthreads();

    for (int t = 0; t < S_; ++t) {
        const int buf = t & 1;

        // ---- stage x tile (double-buffered TDM prefetch of t+1) ----
#if USE_TDM
        if (wave == 0) {
            if (t + 1 < S_) {
                tdm_issue((unsigned)(size_t)&xstage[buf ^ 1][0],
                          xg + (size_t)(t + 1) * E_);
                __builtin_amdgcn_s_wait_tensorcnt((short)1);  // tile t complete
            } else {
                __builtin_amdgcn_s_wait_tensorcnt((short)0);
            }
        }
#else
        {
            const int r = tid >> 4, seg = tid & 15;
            const float4* src = (const float4*)(xg + (size_t)r * S_ * E_ +
                                                (size_t)t * E_ + seg * 8);
            float4 a0 = src[0], a1 = src[1];
            *(float4*)&xstage[buf][r * XLD + seg * 8]     = a0;
            *(float4*)&xstage[buf][r * XLD + seg * 8 + 4] = a1;
        }
#endif
        __syncthreads();   // SYNC_A: x tile + prev h/hk visible

        // ---- gate GEMM: g = x*Wih^T + h*Whh^T + bias (WMMA f16->f32) ----
        const float* xs = xstage[buf];
        v8f acc0 = {0.f,0.f,0.f,0.f,0.f,0.f,0.f,0.f};
        v8f acc1 = {0.f,0.f,0.f,0.f,0.f,0.f,0.f,0.f};
        #pragma unroll
        for (int kc = 0; kc < 4; ++kc) {
            v16h a = gather_a_f32(xs, XLD, lane, kc * 32);
            acc0 = wmma_f16(a, wif[0][kc], acc0);
            acc1 = wmma_f16(a, wif[1][kc], acc1);
        }
        #pragma unroll
        for (int kc = 0; kc < 2; ++kc) {
            v16h a = gather_a_f16(h16_s, HLD16, lane, kc * 32);
            acc0 = wmma_f16(a, whf[0][kc], acc0);
            acc1 = wmma_f16(a, whf[1][kc], acc1);
        }
        {
            const int na = n0 + (lane & 15);
            #pragma unroll
            for (int r = 0; r < 8; ++r) {
                const int m = r + ((lane >> 4) << 3);   // C/D layout: M=r(+8)
                g_s[m * GLD + na]      = acc0[r] + gb[0];
                g_s[m * GLD + na + 16] = acc1[r] + gb[1];
            }
        }
        __syncthreads();   // SYNC_B: gates complete

        // ---- LSTM cell update: 16x64 = 1024 cells, 4 per thread ----
        #pragma unroll
        for (int r = 0; r < 4; ++r) {
            const int idx = tid + (r << 8);
            const int b = idx >> 6, hh = idx & 63;
            const float* gr = g_s + b * GLD;
            const float gi = gr[hh], gf = gr[64 + hh];
            const float gc = gr[128 + hh], go = gr[192 + hh];
            const float cc = sigmoidf_(gf) * c_s[b * H_ + hh] +
                             sigmoidf_(gi) * tanhf(gc);
            c_s[b * H_ + hh] = cc;
            h16_s[b * HLD16 + hh] = (_Float16)(sigmoidf_(go) * tanhf(cc));
        }
        __syncthreads();   // SYNC_C: new h (f16) ready

        // ---- head GEMM on matrix pipes: z = h . hW^T (waves 0-1) ----
        if (wave < 2) {
            v8f hz = {0.f,0.f,0.f,0.f,0.f,0.f,0.f,0.f};
            #pragma unroll
            for (int kc = 0; kc < 2; ++kc) {
                v16h a = gather_a_f16(h16_s, HLD16, lane, kc * 32);
                hz = wmma_f16(a, hwf[kc], hz);
            }
            const int n = wave * 16 + (lane & 15);
            if (n < NH_) {
                #pragma unroll
                for (int r = 0; r < 8; ++r) {
                    const int m = r + ((lane >> 4) << 3);
                    z_s[m * ZLD + n] = hz[r];
                }
            }
        }
        __syncthreads();   // SYNC_D: z ready

        // ---- head recurrence + tanh: 416 items, <=2 per thread ----
        float z0, z1 = 0.f;
        const int it1 = tid + 256;
        const bool has1 = (it1 < NHEAD);
        {
            const int b = tid / NH_, r = tid - b * NH_, k = r >> 1;
            const float* hkp = hk_s + b * NH_ + k * H2_;
            z0 = z_s[b * ZLD + r] + hb_s[r] +
                 hkp[0] * hR_s[r * 2 + 0] + hkp[1] * hR_s[r * 2 + 1];
        }
        if (has1) {
            const int b = it1 / NH_, r = it1 - b * NH_, k = r >> 1;
            const float* hkp = hk_s + b * NH_ + k * H2_;
            z1 = z_s[b * ZLD + r] + hb_s[r] +
                 hkp[0] * hR_s[r * 2 + 0] + hkp[1] * hR_s[r * 2 + 1];
        }
        __syncthreads();   // SYNC_E: old hk fully consumed

        {
            const int b = tid / NH_, r = tid - b * NH_, k = r >> 1, o = r & 1;
            const float th = tanhf(z0);
            hk_s[b * NH_ + k * H2_ + o] = th;
            if (t == len_s[b] - 1)
                out[((size_t)k * B_ + (bbase + b)) * H2_ + o] = sigmoidf_(th);
        }
        if (has1) {
            const int b = it1 / NH_, r = it1 - b * NH_, k = r >> 1, o = r & 1;
            const float th = tanhf(z1);
            hk_s[b * NH_ + k * H2_ + o] = th;
            if (t == len_s[b] - 1)
                out[((size_t)k * B_ + (bbase + b)) * H2_ + o] = sigmoidf_(th);
        }
    }
}

extern "C" void kernel_launch(void* const* d_in, const int* in_sizes, int n_in,
                              void* d_out, int out_size, void* d_ws, size_t ws_size,
                              hipStream_t stream) {
    (void)in_sizes; (void)n_in; (void)out_size; (void)d_ws; (void)ws_size;
    const float*     input   = (const float*)d_in[0];
    const long long* lengths = (const long long*)d_in[1];
    const float*     Wih     = (const float*)d_in[2];
    const float*     Whh     = (const float*)d_in[3];
    const float*     bih     = (const float*)d_in[4];
    const float*     bhh     = (const float*)d_in[5];
    const float*     hWih    = (const float*)d_in[6];
    const float*     hWhh    = (const float*)d_in[7];
    const float*     hbih    = (const float*)d_in[8];
    const float*     hbhh    = (const float*)d_in[9];
    float*           out     = (float*)d_out;

    lstm_head_persist<<<NWG, 256, 0, stream>>>(
        input, lengths, Wih, Whh, bih, bhh, hWih, hWhh, hbih, hbhh, out);
}
